// CIFModule_27006754357837
// MI455X (gfx1250) — matrix operand order repaired and checked by reading, not compiled, not run on hardware
//
#include <hip/hip_runtime.h>
#include <hip/hip_bf16.h>
#include <math.h>

// ---------------------------------------------------------------------------
// CIF forward for MI455X (gfx1250, wave32, WMMA), round 2.
// Fragment-major LDS/global swizzles so every WMMA operand is loaded with
// 2x b128 instead of 16x scalar u16 + b16 packing.
// ---------------------------------------------------------------------------

typedef __attribute__((ext_vector_type(16))) _Float16 v16h;
typedef __attribute__((ext_vector_type(8)))  _Float16 h8;
typedef __attribute__((ext_vector_type(4)))  _Float16 h4;
typedef __attribute__((ext_vector_type(8)))  float    v8f;

#define B_     32
#define T_     3000
#define D_     128
#define TSW    375
#define DSW    192
#define TS1    188
#define DS1    192
#define NMAX   128
#define HID_   128
#define DMODEL 512
#define CATD   512

// output layout: embs [B,N,512] | alpha [B,T] | qty (1)
#define EMB_OFF   0
#define ALPHA_OFF (B_ * NMAX * DMODEL)        // 2097152
#define QTY_OFF   (ALPHA_OFF + B_ * T_)       // 2193152

// workspace layout (bytes)
#define WS_PROJ_OFF  0                                 // 512*512 f16 swizzled
#define WS_ONSET_OFF (DMODEL * CATD * 2)               // 32 floats
#define WS_FF_OFF    (WS_ONSET_OFF + 32 * 4)           // 4096 ints
#define WS_NF_OFF    (WS_FF_OFF + B_ * NMAX * 4)       // 32 ints

// A-fragment swizzle (16-bit A 16x32, ISA 7.12.2):
//   element (m, klocal) lives in lane = m + 16*((klocal>>3)&1),
//   half    h = (klocal&7) + ((klocal>>4)&1)*8
// B-fragment swizzle (16-bit B 32x16):
//   element (klocal, n) lives in lane = (n&15) + 16*((klocal>>3)&1), half h = (klocal&7) + ((klocal>>4)&1)*8

__device__ __forceinline__ v16h frag_cat(h8 lo, h8 hi) {
    return __builtin_shufflevector(lo, hi, 0,1,2,3,4,5,6,7,8,9,10,11,12,13,14,15);
}

__global__ __launch_bounds__(64)
void cif_init(float* __restrict__ onset, float* __restrict__ qty) {
    int tid = threadIdx.x;
    if (tid < B_) onset[tid] = 0.0f;
    if (tid == 32) *qty = 0.0f;
}

// ---------------------------------------------------------------------------
// Pre-swizzle proj_w (512x512 f32) into fragment-major f16 in workspace:
// [nch(4)][nt(8)][kt(16)][lane(32)][h(16)]
// ---------------------------------------------------------------------------
__global__ __launch_bounds__(256)
void cif_swizzle_proj(const float* __restrict__ proj_w, _Float16* __restrict__ wsB) {
    int idx = blockIdx.x * 256 + threadIdx.x;      // over 512*512
    int k = idx >> 9;
    int n = idx & 511;
    int nch = n >> 7, nn = n & 127, nt = nn >> 4, lrow = nn & 15;
    int kt = k >> 5, klocal = k & 31, halfB = (klocal >> 3) & 1, h = (klocal & 7) + ((klocal >> 4) & 1) * 8;
    int lane = lrow + 16 * halfB;
    wsB[((((nch * 8 + nt) * 16 + kt) * 32) + lane) * 16 + h] = (_Float16)proj_w[idx];
}

// ---------------------------------------------------------------------------
// Kernel 1: weight predictor. 128 threads = 4 waves, 16 rows (b,t) per wave.
// ---------------------------------------------------------------------------
__global__ __launch_bounds__(128)
void cif_predictor(const float* __restrict__ fs,
                   const int*   __restrict__ in_len,
                   const float* __restrict__ conv_w,
                   const float* __restrict__ ln_g,
                   const float* __restrict__ ln_b,
                   const float* __restrict__ wd_w,
                   const float* __restrict__ wd_b,
                   const float* __restrict__ wp_w,
                   const float* __restrict__ wp_b,
                   float* __restrict__ alpha_out,
                   float* __restrict__ onset_sum)
{
    // fragment-major tiles: per-lane 16 contiguous halves -> 2x ds_load_b128
    __shared__ _Float16 sAfrag[4][4][32][16] __attribute__((aligned(16)));  // [wave][kt][lane][h]
    __shared__ _Float16 sBfrag[8][4][32][16] __attribute__((aligned(16)));  // [nt][kt][lane][h]
    __shared__ float sCw[D_ * 3];
    __shared__ float sLg[D_], sLb[D_], sWp[HID_], sWb[HID_];

    const int tid  = threadIdx.x;
    const int lane = tid & 31;
    const int wave = tid >> 5;
    const int half = lane >> 4;
    const int lrow = lane & 15;

    for (int i = tid; i < D_ * 3; i += 128) sCw[i] = conv_w[i];
    for (int i = tid; i < D_; i += 128) {
        sLg[i] = ln_g[i]; sLb[i] = ln_b[i];
        sWp[i] = wp_w[i]; sWb[i] = wd_b[i];
    }
    for (int i = tid; i < HID_ * D_; i += 128) {
        int k = i >> 7, n = i & 127;
        int nt = n >> 4, lr = n & 15;
        int kt = k >> 5, klocal = k & 31, hb = (klocal >> 3) & 1, h = (klocal & 7) + ((klocal >> 4) & 1) * 8;
        sBfrag[nt][kt][lr + 16 * hb][h] = (_Float16)wd_w[i];
    }

    const int rowBase = blockIdx.x * 64 + wave * 16;

    // phase 1: depthwise conv + residual + LayerNorm -> swizzled f16 A tile
    const int d0 = lane * 4;                    // 4 contiguous channels per lane
    const int ktA     = d0 >> 5;
    const int klA     = d0 & 31;
    const int halfA   = (klA >> 3) & 1;
    const int hA      = (klA & 7) + ((klA >> 4) & 1) * 8;
    for (int m = 0; m < 16; ++m) {
        int grow = rowBase + m;
        int b = grow / T_;
        int t = grow - b * T_;
        const float* base = fs + ((long)b * T_ + t) * D_;
        __builtin_prefetch(base + 16 * D_, 0, 0);        // stream next tile
        float4 c1 = *(const float4*)(base + d0);
        float4 c0 = (t > 0)      ? *(const float4*)(base + d0 - D_) : make_float4(0.f,0.f,0.f,0.f);
        float4 c2 = (t < T_ - 1) ? *(const float4*)(base + d0 + D_) : make_float4(0.f,0.f,0.f,0.f);
        const float* p0 = (const float*)&c0;
        const float* p1 = (const float*)&c1;
        const float* p2 = (const float*)&c2;
        float vals[4], s = 0.f, s2 = 0.f;
        #pragma unroll
        for (int i = 0; i < 4; ++i) {
            int d = d0 + i;
            float v = fmaf(p0[i], sCw[d*3+0],
                      fmaf(p1[i], sCw[d*3+1],
                      fmaf(p2[i], sCw[d*3+2], p1[i])));
            vals[i] = v; s += v; s2 += v * v;
        }
        #pragma unroll
        for (int o = 16; o > 0; o >>= 1) {
            s  += __shfl_xor(s,  o, 32);
            s2 += __shfl_xor(s2, o, 32);
        }
        float mu   = s  * (1.f / 128.f);
        float var  = s2 * (1.f / 128.f) - mu * mu;
        float rstd = rsqrtf(var + 1e-5f);
        h4 pack;
        #pragma unroll
        for (int i = 0; i < 4; ++i) {
            int d = d0 + i;
            pack[i] = (_Float16)((vals[i] - mu) * rstd * sLg[d] + sLb[d]);
        }
        *(h4*)&sAfrag[wave][ktA][m + 16 * halfA][hA] = pack;   // ds_store_b64
    }
    __syncthreads();

    // phase 2: h = relu(x @ wd_w + wd_b); alpha_acc = h @ wp_w  (fused)
    float aAcc[8];
    #pragma unroll
    for (int v = 0; v < 8; ++v) aAcc[v] = 0.f;

    for (int nt = 0; nt < 8; ++nt) {
        v8f acc = {0.f,0.f,0.f,0.f,0.f,0.f,0.f,0.f};
        #pragma unroll
        for (int kt = 0; kt < 4; ++kt) {
            h8 alo = *(const h8*)&sAfrag[wave][kt][lane][0];   // ds_load_b128
            h8 ahi = *(const h8*)&sAfrag[wave][kt][lane][8];
            h8 blo = *(const h8*)&sBfrag[nt][kt][lane][0];
            h8 bhi = *(const h8*)&sBfrag[nt][kt][lane][8];
            acc = __builtin_amdgcn_wmma_f32_16x16x32_f16(
                      false, frag_cat(alo, ahi), false, frag_cat(blo, bhi),
                      (short)0, acc, false, false);
        }
        int n = nt * 16 + lrow;
        float wpn = sWp[n], wbn = sWb[n];
        #pragma unroll
        for (int v = 0; v < 8; ++v) {
            float hv = fmaxf(acc[v] + wbn, 0.f);
            aAcc[v] += hv * wpn;
        }
    }
    #pragma unroll
    for (int v = 0; v < 8; ++v) {
        #pragma unroll
        for (int o = 8; o > 0; o >>= 1) aAcc[v] += __shfl_xor(aAcc[v], o, 32);
    }

    if (lrow == 0) {   // lanes 0 (m=v) and 16 (m=v+8)
        float wpb = wp_b[0];
        #pragma unroll
        for (int v = 0; v < 8; ++v) {
            int m = v + half * 8;
            int grow = rowBase + m;
            int b = grow / T_;
            int t = grow - b * T_;
            float z = aAcc[v] + wpb;
            float alpha = fmaxf(z, 0.f) + log1pf(expf(-fabsf(z)));   // softplus
            float onset = 1.f / (1.f + expf(-(alpha - 1.0f) * 10.0f));
            if (t >= in_len[b]) { alpha = 0.f; onset = 0.f; }
            alpha_out[b * T_ + t] = alpha;
            atomicAdd(onset_sum + b, onset);
        }
    }
}

// ---------------------------------------------------------------------------
// Kernel 2: per-batch scan -> fire_frames + nfires + qty. 32 blocks x 256.
// ---------------------------------------------------------------------------
__global__ __launch_bounds__(256)
void cif_scan(const float* __restrict__ alpha,
              const float* __restrict__ onset_sum,
              const int*   __restrict__ target_len,
              int*   __restrict__ ff,
              int*   __restrict__ nfires,
              float* __restrict__ qty)
{
    __shared__ int tot[256];
    const int b   = blockIdx.x;
    const int tid = threadIdx.x;

    if (tid < NMAX) ff[b * NMAX + tid] = T_ - 1;   // overflow slots -> clip(T)

    const float* a  = alpha + b * T_;
    const int    t0 = tid * 12;                    // 250*12 = 3000 exactly
    int mask[12]; int cnt = 0;
    if (t0 < T_) {                                 // 16B-aligned float4 reads
        float4 v0 = *(const float4*)(a + t0);
        float4 v1 = *(const float4*)(a + t0 + 4);
        float4 v2 = *(const float4*)(a + t0 + 8);
        const float* pv = (const float*)&v0;
        #pragma unroll
        for (int j = 0; j < 4; ++j) { mask[j]   = pv[j] > 1.0f; cnt += mask[j]; }
        pv = (const float*)&v1;
        #pragma unroll
        for (int j = 0; j < 4; ++j) { mask[4+j] = pv[j] > 1.0f; cnt += mask[4+j]; }
        pv = (const float*)&v2;
        #pragma unroll
        for (int j = 0; j < 4; ++j) { mask[8+j] = pv[j] > 1.0f; cnt += mask[8+j]; }
    } else {
        #pragma unroll
        for (int j = 0; j < 12; ++j) mask[j] = 0;
    }
    tot[tid] = cnt;
    __syncthreads();
    for (int off = 1; off < 256; off <<= 1) {      // Hillis-Steele inclusive
        int v = (tid >= off) ? tot[tid - off] : 0;
        __syncthreads();
        tot[tid] += v;
        __syncthreads();
    }
    int c = tot[tid] - cnt;                        // exclusive prefix
    for (int j = 0; j < 12; ++j) {
        if (mask[j]) {
            ++c;
            if (c <= NMAX) ff[b * NMAX + (c - 1)] = t0 + j;
        }
    }
    if (tid == 0) {
        int total = tot[255];
        nfires[b] = total < NMAX ? total : NMAX;
        atomicAdd(qty, fabsf(onset_sum[b] - (float)target_len[b]) * (1.0f / B_));
    }
}

// ---------------------------------------------------------------------------
// Kernel 3: gather cat=[temporal|pitch|beat], WMMA 512->512 projection.
// 128 blocks x 64 threads (2 waves); 32 rows per block, 16 per wave.
// ---------------------------------------------------------------------------
__global__ __launch_bounds__(64)
void cif_output(const float* __restrict__ fs,
                const float* __restrict__ asrc,
                const float* __restrict__ s1,
                const int*   __restrict__ ff,
                const int*   __restrict__ nfires,
                const _Float16* __restrict__ wsB,   // pre-swizzled proj_w
                const float* __restrict__ proj_b,
                float* __restrict__ embs)
{
    __shared__ _Float16 sCatFrag[2][16][32][16] __attribute__((aligned(16))); // [wave][kt][lane][h]
    __shared__ int sFF[32];
    __shared__ int sTV[32];

    const int tid  = threadIdx.x;
    const int lane = tid & 31;
    const int wave = tid >> 5;
    const int half = lane >> 4;
    const int lrow = lane & 15;
    const int rowBase = blockIdx.x * 32;

    if (tid < 32) {
        int gr = rowBase + tid;
        int bb = gr >> 7;
        int n  = gr & 127;
        sFF[tid] = ff[gr];
        sTV[tid] = (n < nfires[bb]) ? 1 : 0;
    }
    __syncthreads();

    // gather + swizzle: 4 channels per step (groups of 4 never straddle
    // the 128/320 boundaries nor the 8-half sub-blocks)
    for (int idx = tid * 4; idx < 32 * CATD; idx += 64 * 4) {
        int r  = idx >> 9;
        int d  = idx & 511;
        int gr = rowBase + r;
        int bb = gr >> 7;
        int f  = sFF[r];
        float vv[4];
        if (d < 128) {
            if (sTV[r]) {
                float4 v = *(const float4*)(fs + ((long)bb * T_ + f) * D_ + d);
                const float* pv = (const float*)&v;
                #pragma unroll
                for (int i = 0; i < 4; ++i) vv[i] = pv[i];
            } else {
                #pragma unroll
                for (int i = 0; i < 4; ++i) vv[i] = 0.f;
            }
        } else if (d < 320) {
            int f0 = (int)((float)f * (375.0f / 3000.0f));
            f0 = f0 > TSW - 1 ? TSW - 1 : f0;
            float4 v = *(const float4*)(asrc + ((long)bb * TSW + f0) * DSW + (d - 128));
            const float* pv = (const float*)&v;
            #pragma unroll
            for (int i = 0; i < 4; ++i) vv[i] = pv[i];
        } else {
            int f1 = (int)((float)f * (188.0f / 3000.0f));
            f1 = f1 > TS1 - 1 ? TS1 - 1 : f1;
            float4 v = *(const float4*)(s1 + ((long)bb * TS1 + f1) * DS1 + (d - 320));
            const float* pv = (const float*)&v;
            #pragma unroll
            for (int i = 0; i < 4; ++i) vv[i] = pv[i];
        }
        int m = r & 15, w = r >> 4;
        int kt = d >> 5, klocal = d & 31;
        int halfA = (klocal >> 3) & 1;
        int h = (klocal & 7) + ((klocal >> 4) & 1) * 8;
        h4 pack;
        #pragma unroll
        for (int i = 0; i < 4; ++i) pack[i] = (_Float16)vv[i];
        *(h4*)&sCatFrag[w][kt][m + 16 * halfA][h] = pack;      // ds_store_b64
    }
    __syncthreads();

    for (int nch = 0; nch < 4; ++nch) {            // 4 chunks of N=128
        v8f acc[8];
        #pragma unroll
        for (int nt = 0; nt < 8; ++nt)
            acc[nt] = (v8f){0.f,0.f,0.f,0.f,0.f,0.f,0.f,0.f};
        for (int kt = 0; kt < 16; ++kt) {          // K = 512 in steps of 32
            h8 alo = *(const h8*)&sCatFrag[wave][kt][lane][0]; // ds_load_b128
            h8 ahi = *(const h8*)&sCatFrag[wave][kt][lane][8];
            v16h af = frag_cat(alo, ahi);
            #pragma unroll
            for (int nt = 0; nt < 8; ++nt) {
                const h8* pb = (const h8*)(wsB +
                    (size_t)((((nch * 8 + nt) * 16 + kt) * 32) + lane) * 16);
                h8 blo = pb[0];                                // global_load_b128
                h8 bhi = pb[1];
                acc[nt] = __builtin_amdgcn_wmma_f32_16x16x32_f16(
                              false, af, false, frag_cat(blo, bhi),
                              (short)0, acc[nt], false, false);
            }
        }
        #pragma unroll
        for (int nt = 0; nt < 8; ++nt) {
            int col = nch * 128 + nt * 16 + lrow;
            float pb = proj_b[col];
            #pragma unroll
            for (int v = 0; v < 8; ++v) {
                int m = v + half * 8;
                int grow = rowBase + wave * 16 + m;
                embs[(long)grow * DMODEL + col] = acc[nt][v] + pb;
            }
        }
    }
}

// ---------------------------------------------------------------------------
extern "C" void kernel_launch(void* const* d_in, const int* in_sizes, int n_in,
                              void* d_out, int out_size, void* d_ws, size_t ws_size,
                              hipStream_t stream)
{
    const float* fs      = (const float*)d_in[0];
    const float* asrc    = (const float*)d_in[1];
    const float* s1      = (const float*)d_in[2];
    const int*   in_len  = (const int*)  d_in[3];
    const int*   tgt_len = (const int*)  d_in[4];
    const float* conv_w  = (const float*)d_in[5];
    const float* ln_g    = (const float*)d_in[6];
    const float* ln_b    = (const float*)d_in[7];
    const float* wd_w    = (const float*)d_in[8];
    const float* wd_b    = (const float*)d_in[9];
    const float* wp_w    = (const float*)d_in[10];
    const float* wp_b    = (const float*)d_in[11];
    const float* proj_w  = (const float*)d_in[12];
    const float* proj_b  = (const float*)d_in[13];

    float* out   = (float*)d_out;
    float* embs  = out + EMB_OFF;
    float* alpha = out + ALPHA_OFF;
    float* qty   = out + QTY_OFF;

    char*      ws       = (char*)d_ws;
    _Float16*  ws_projB = (_Float16*)(ws + WS_PROJ_OFF);
    float*     ws_onset = (float*)(ws + WS_ONSET_OFF);
    int*       ws_ff    = (int*)  (ws + WS_FF_OFF);
    int*       ws_nf    = (int*)  (ws + WS_NF_OFF);

    cif_init<<<1, 64, 0, stream>>>(ws_onset, qty);
    cif_swizzle_proj<<<(DMODEL * CATD) / 256, 256, 0, stream>>>(proj_w, ws_projB);
    cif_predictor<<<(B_ * T_) / 64, 128, 0, stream>>>(
        fs, in_len, conv_w, ln_g, ln_b, wd_w, wd_b, wp_w, wp_b, alpha, ws_onset);
    cif_scan<<<B_, 256, 0, stream>>>(alpha, ws_onset, tgt_len, ws_ff, ws_nf, qty);
    cif_output<<<(B_ * NMAX) / 32, 64, 0, stream>>>(
        fs, asrc, s1, ws_ff, ws_nf, ws_projB, proj_b, embs);
}
